// AreaAttentionBlock_65180423685282
// MI455X (gfx1250) — matrix operand order repaired and checked by reading, not compiled, not run on hardware
//
#include <hip/hip_runtime.h>
#include <hip/hip_bf16.h>
#include <stdint.h>

// ---- problem constants ----
#define BB    2
#define CCH   256
#define NN    4096          // H*W = 64*64
#define NT    8192          // BB*NN (batched columns)
#define NHEAD 8
#define HD    32
#define AREA  4
#define NAr   1024          // NN/AREA
#define QKC   512           // 2*C
#define MQKV  768           // 512 + 256 stacked
#define MD    307           // int(1.2*256)
#define MDP   320           // padded to multiple of 64

typedef __bf16 bf16;
typedef __attribute__((ext_vector_type(16))) __bf16 v16bf;
typedef __attribute__((ext_vector_type(8)))  float  v8f;

// ---------------- helpers ----------------
static __device__ __forceinline__ unsigned short bfbits(float f) {
  union { float f; unsigned u; } a; a.f = f;
  return (unsigned short)((a.u + 0x7FFFu + ((a.u >> 16) & 1u)) >> 16);  // RNE
}
static __device__ __forceinline__ bf16 f2bf(float f) {
  union { unsigned short s; bf16 b; } o; o.s = bfbits(f); return o.b;
}
static __device__ __forceinline__ float bf2f(bf16 b) {
  union { unsigned short s; bf16 b; } i; i.b = b;
  union { unsigned u; float f; } o; o.u = ((unsigned)i.s) << 16; return o.f;
}
static __device__ __forceinline__ v8f vzero() {
  v8f z = {0.f,0.f,0.f,0.f,0.f,0.f,0.f,0.f}; return z;
}
static __device__ __forceinline__ v8f wmma_bf16(v16bf a, v16bf b, v8f c) {
  // D = A(16x32 bf16) x B(32x16 bf16) + C(16x16 f32)
  return __builtin_amdgcn_wmma_f32_16x16x32_bf16(false, a, false, b, (short)0, c, false, false);
}
// cross-half (xor 16) combine via LDS swizzle hardware (no LDS memory touched)
static __device__ __forceinline__ float swz16(float x) {
  return __int_as_float(__builtin_amdgcn_ds_swizzle(__float_as_int(x), 0x401F)); // SWAPX16
}

// CDNA5 async DMA: 16B global -> LDS, tracked by ASYNCcnt (no VGPR staging)
static __device__ __forceinline__ void async_copy16(const bf16* smem_dst,
                                                    const bf16* gsrc) {
  unsigned lds_off = (unsigned)(size_t)smem_dst;     // low 32b of shared aperture addr
  unsigned long long ga = (unsigned long long)(size_t)gsrc;
  asm volatile("global_load_async_to_lds_b128 %0, %1, off"
               :: "v"(lds_off), "v"(ga) : "memory");
}
static __device__ __forceinline__ void async_wait_all() {
  asm volatile("s_wait_asynccnt 0" ::: "memory");
}

// A-fragment: 16x32 bf16, lane r = row, element e -> k = (e&7) + (e>>3)*16 + half*8
static __device__ __forceinline__ v16bf load_fragA(const bf16* base,
                                                   int row0, int ld, int k0, int lane) {
  const int half = lane >> 4, r = lane & 15;
  const bf16* p = base + (size_t)(row0 + r) * ld + k0 + half * 8;
  union { uint4 u[2]; v16bf v; } f;
  f.u[0] = *(const uint4*)(p);        // k = k0+half*8   .. +7
  f.u[1] = *(const uint4*)(p + 16);   // k = k0+16+half*8 .. +7
  return f.v;
}
// B-fragment: 32x16 bf16, lane r = column, element e -> k = (e&7) + (e>>3)*16 + half*8
static __device__ __forceinline__ v16bf load_fragB(const bf16* base,
                                                   int col0, int ld, int k0, int lane) {
  const int half = lane >> 4, r = lane & 15;
  const bf16* p = base + (size_t)(col0 + r) * ld + k0 + half * 8;
  union { uint4 u[2]; v16bf v; } f;
  f.u[0] = *(const uint4*)(p);
  f.u[1] = *(const uint4*)(p + 16);
  return f.v;
}

// ---------------- weight convert + pad to bf16 ----------------
__global__ __launch_bounds__(256) void cvt_pad(const float* __restrict__ src,
                                               bf16* __restrict__ dst,
                                               int Mp, int Kp, int Ms, int Ks) {
  int idx = blockIdx.x * 256 + threadIdx.x;
  if (idx >= Mp * Kp) return;
  int row = idx / Kp, col = idx - row * Kp;
  float v = (row < Ms && col < Ks) ? src[(size_t)row * Ks + col] : 0.f;
  dst[idx] = f2bf(v);
}

// ---------------- x (f32 [b][c][n]) -> xt (bf16 [b][n][c]) ----------------
__global__ __launch_bounds__(256) void transpose_cvt(const float* __restrict__ x,
                                                     bf16* __restrict__ xt) {
  __shared__ float tile[32][33];
  int b = blockIdx.z, c0 = blockIdx.y * 32, n0 = blockIdx.x * 32;
  int tx = threadIdx.x, ty = threadIdx.y;           // 32 x 8
  #pragma unroll
  for (int i = 0; i < 32; i += 8)
    tile[ty + i][tx] = x[(size_t)b * CCH * NN + (size_t)(c0 + ty + i) * NN + n0 + tx];
  __syncthreads();
  #pragma unroll
  for (int i = 0; i < 32; i += 8)
    xt[(size_t)(b * NN + n0 + ty + i) * CCH + c0 + tx] = f2bf(tile[tx][ty + i]);
}

// ------- shared GEMM core: block tile 64x128, async-LDS staged, wave = 32x32 -------
static __device__ __forceinline__ void gemm_core(const bf16* __restrict__ A,
                                                 const bf16* __restrict__ Bm,
                                                 int K, int bm0, int bn0,
                                                 int wave, int lane,
                                                 v8f (&acc)[2][2]) {
  __shared__ bf16 sA[64 * 32];     // A block tile, row-major [m][k]
  __shared__ bf16 sB[128 * 32];    // B block tile, row-major [n][k]
  const int t = wave * 32 + lane;  // 0..255
  const int wm = wave & 1, wn = wave >> 1;
  const int row = t >> 1, part = (t & 1) * 16;   // one 16B chunk each

  for (int k0 = 0; k0 < K; k0 += 32) {
    // stage A (64x32 = 4KB: 1 chunk/thread) and B (128x32 = 8KB: 2 chunks/thread)
    async_copy16(&sA[row * 32 + part],        A  + (size_t)(bm0 + row) * K + k0 + part);
    async_copy16(&sB[row * 32 + part],        Bm + (size_t)(bn0 + row) * K + k0 + part);
    async_copy16(&sB[(row + 64) * 32 + part], Bm + (size_t)(bn0 + row + 64) * K + k0 + part);
    async_wait_all();
    __syncthreads();

    v16bf a0 = load_fragA(sA, wm * 32,      32, 0, lane);
    v16bf a1 = load_fragA(sA, wm * 32 + 16, 32, 0, lane);
    v16bf b0 = load_fragB(sB, wn * 32,      32, 0, lane);
    v16bf b1 = load_fragB(sB, wn * 32 + 16, 32, 0, lane);
    acc[0][0] = wmma_bf16(a0, b0, acc[0][0]);
    acc[0][1] = wmma_bf16(a0, b1, acc[0][1]);
    acc[1][0] = wmma_bf16(a1, b0, acc[1][0]);
    acc[1][1] = wmma_bf16(a1, b1, acc[1][1]);
    __syncthreads();                       // protect sA/sB before next stage
  }
}
#define GEMM_PROLOG                                              \
  int lane = threadIdx.x & 31, wave = threadIdx.x >> 5;          \
  int bm0 = blockIdx.y * 64, bn0 = blockIdx.x * 128;             \
  int m0 = bm0 + (wave & 1) * 32;                                \
  int n0 = bn0 + (wave >> 1) * 32;                               \
  v8f acc[2][2];                                                 \
  acc[0][0] = vzero(); acc[0][1] = vzero();                      \
  acc[1][0] = vzero(); acc[1][1] = vzero();

// ---------------- GEMM 1: [w_qk;w_v](768x256) x X -> q,k ([n][512]) and v ([c][n]) ----------------
__global__ __launch_bounds__(256) void gemm_qkv(const bf16* __restrict__ wqkv,
                                                const bf16* __restrict__ xt,
                                                const float* __restrict__ s_qk,
                                                const float* __restrict__ b_qk,
                                                const float* __restrict__ s_v,
                                                const float* __restrict__ b_v,
                                                bf16* __restrict__ qkt,
                                                bf16* __restrict__ vbf) {
  GEMM_PROLOG
  gemm_core(wqkv, xt, CCH, bm0, bn0, wave, lane, acc);
  int half = lane >> 4, r = lane & 15;
  #pragma unroll
  for (int nt = 0; nt < 2; nt++) {
    int ng = n0 + nt * 16 + r;               // global column (b*4096+n)
    int bb = ng >> 12, nl = ng & 4095;
    #pragma unroll
    for (int mt = 0; mt < 2; mt++) {
      int ob = m0 + mt * 16 + half * 8;      // first of 8 consecutive output rows
      if (ob < QKC) {                        // q (o<256, fold attn scale) or k
        float sc = (ob < CCH) ? 0.17677669529663687f : 1.0f;  // hd^-0.5
        union { unsigned short s[8]; uint4 u; } pk;
        #pragma unroll
        for (int e = 0; e < 8; e++) {
          int o = ob + e;
          pk.s[e] = bfbits((acc[mt][nt][e] * s_qk[o] + b_qk[o]) * sc);
        }
        *(uint4*)(qkt + (size_t)ng * QKC + ob) = pk.u;
      } else {                               // v: store channel-major [b][c][n]
        #pragma unroll
        for (int e = 0; e < 8; e++) {
          int c = ob + e - QKC;
          vbf[(size_t)bb * CCH * NN + (size_t)c * NN + nl] =
              f2bf(acc[mt][nt][e] * s_v[c] + b_v[c]);
        }
      }
    }
  }
}

// ---------------- attention: K/V panels async-staged to LDS, two-pass softmax ----------------
// dynamic LDS: sK = 1024x32 bf16 (64KB), sV = 32x1024 bf16 (64KB)
__global__ __launch_bounds__(256) void attn_kernel(const bf16* __restrict__ qkt,
                                                   const bf16* __restrict__ vbf,
                                                   bf16* __restrict__ outbf) {
  extern __shared__ bf16 asmem[];
  bf16* sK = asmem;                 // [m][d]  1024 x 32
  bf16* sV = asmem + 1024 * 32;     // [d][m]  32 x 1024

  int lane = threadIdx.x & 31, wave = threadIdx.x >> 5;
  int tid = threadIdx.x;
  int t = blockIdx.x;
  int nblk = t & 7, h = (t >> 3) & 7, ba = t >> 6;
  int b = ba >> 2, area = ba & 3;
  int n0 = nblk * 128 + wave * 16;           // query-tile base within area

  const bf16* qbase = qkt + ((size_t)(b * NN + area * NAr)) * QKC + h * HD;
  const bf16* kbase = qbase + CCH;
  const bf16* vbase = vbf + (size_t)b * CCH * NN + (size_t)(h * HD) * NN + area * NAr;

  // ---- stage whole K panel (1024 rows x 64B) and V panel (32 rows x 2KB) ----
  #pragma unroll
  for (int i = 0; i < 16; i++) {             // K: 4096 16B-chunks / 256 threads
    int c = tid + i * 256;
    int kr = c >> 2, kp = (c & 3) * 8;       // row, elem offset
    async_copy16(&sK[kr * 32 + kp], kbase + (size_t)kr * QKC + kp);
  }
  #pragma unroll
  for (int i = 0; i < 16; i++) {             // V: 4096 16B-chunks / 256 threads
    int c = tid + i * 256;
    int vr = c >> 7, vp = (c & 127) * 8;
    async_copy16(&sV[vr * 1024 + vp], vbase + (size_t)vr * NN + vp);
  }

  // Q fragment (B operand of S^T wmma): rows d contiguous in qkt[n][c]
  v16bf qf = load_fragB(qbase, n0, QKC, 0, lane);

  async_wait_all();
  __syncthreads();

  // ---- pass 1: per-row max (each lane owns its half of one row's scores) ----
  float mrun = -3.0e38f;
  for (int m0 = 0; m0 < NAr; m0 += 32) {
    v16bf k0f = load_fragA(sK, m0,      32, 0, lane);
    v16bf k1f = load_fragA(sK, m0 + 16, 32, 0, lane);
    v8f s0 = wmma_bf16(k0f, qf, vzero());    // S^T tile, m rows m0..m0+15
    v8f s1 = wmma_bf16(k1f, qf, vzero());    // S^T tile, m rows m0+16..m0+31
    #pragma unroll
    for (int e = 0; e < 8; e++) mrun = fmaxf(mrun, fmaxf(s0[e], s1[e]));
  }
  float M = fmaxf(mrun, swz16(mrun));        // combine the two half-lane subsets

  // ---- pass 2: recompute scores, exp, accumulate O and l ----
  const float L2E = 1.44269504f;
  float l = 0.f;
  v8f o0 = vzero(), o1 = vzero();
  for (int m0 = 0; m0 < NAr; m0 += 32) {
    v16bf k0f = load_fragA(sK, m0,      32, 0, lane);
    v16bf k1f = load_fragA(sK, m0 + 16, 32, 0, lane);
    v8f s0 = wmma_bf16(k0f, qf, vzero());
    v8f s1 = wmma_bf16(k1f, qf, vzero());
    // S^T D-layout == P A-fragment layout: zero cross-lane data movement
    union { unsigned short s[16]; v16bf v; } pf;
    #pragma unroll
    for (int e = 0; e < 8; e++) {
      float p0 = exp2f((s0[e] - M) * L2E);
      float p1 = exp2f((s1[e] - M) * L2E);
      l += p0 + p1;
      pf.s[e]     = bfbits(p0);              // k = m (e&7)+half*8
      pf.s[8 + e] = bfbits(p1);              // k = m 16+(e&7)+half*8
    }
    v16bf v0f = load_fragB(sV, 0,  1024, m0, lane);   // d cols 0..15
    v16bf v1f = load_fragB(sV, 16, 1024, m0, lane);   // d cols 16..31
    o0 = wmma_bf16(pf.v, v0f, o0);
    o1 = wmma_bf16(pf.v, v1f, o1);
  }
  float rcp = 1.0f / (l + swz16(l));         // 1/l for row n = lane&15

  // ---- store O * (1/l): O tile lane = d col, element e = row half*8+e ----
  int half = lane >> 4, r = lane & 15;
  size_t obase = ((size_t)(b * NN + area * NAr + n0)) * CCH + h * HD;
  #pragma unroll
  for (int e = 0; e < 8; e++) {
    int nrow = half * 8 + e;
    float rn = __int_as_float(
        __builtin_amdgcn_ds_bpermute(nrow << 2, __float_as_int(rcp)));
    size_t rowoff = obase + (size_t)nrow * CCH;
    outbf[rowoff + r]      = f2bf(o0[e] * rn);
    outbf[rowoff + 16 + r] = f2bf(o1[e] * rn);
  }
}

// ---------------- depthwise 3x3 PE, added in-place into outbf [b][n][c] ----------------
__global__ __launch_bounds__(256) void dwconv_pe(const bf16* __restrict__ vbf,
                                                 const float* __restrict__ w_pe,
                                                 const float* __restrict__ s_pe,
                                                 const float* __restrict__ b_pe,
                                                 bf16* __restrict__ outbf) {
  int idx = blockIdx.x * 256 + threadIdx.x;
  if (idx >= BB * CCH * NN) return;
  int n = idx & 4095, c = (idx >> 12) & 255, b = idx >> 20;
  int hh = n >> 6, ww = n & 63;
  const bf16* vp = vbf + (size_t)b * CCH * NN + (size_t)c * NN;
  const float* wp = w_pe + c * 9;
  float acc = 0.f;
  #pragma unroll
  for (int di = -1; di <= 1; di++) {
    int h2 = hh + di;
    if (h2 < 0 || h2 > 63) continue;
    #pragma unroll
    for (int dj = -1; dj <= 1; dj++) {
      int w2 = ww + dj;
      if (w2 < 0 || w2 > 63) continue;
      acc += bf2f(vp[h2 * 64 + w2]) * wp[(di + 1) * 3 + (dj + 1)];
    }
  }
  float pe = acc * s_pe[c] + b_pe[c];
  size_t oi = ((size_t)(b * NN + n)) * CCH + c;
  outbf[oi] = f2bf(bf2f(outbf[oi]) + pe);
}

// ---------------- proj GEMM: x1 = x + s*W_proj*(out+pe)+b ; emit f32 [c][n] and bf16 [n][c] ----------------
__global__ __launch_bounds__(256) void gemm_proj(const bf16* __restrict__ wproj,
                                                 const bf16* __restrict__ outbf,
                                                 const float* __restrict__ s_p,
                                                 const float* __restrict__ b_p,
                                                 const float* __restrict__ x,
                                                 float* __restrict__ x1f,
                                                 bf16* __restrict__ x1bf) {
  GEMM_PROLOG
  gemm_core(wproj, outbf, CCH, bm0, bn0, wave, lane, acc);
  int half = lane >> 4, r = lane & 15;
  #pragma unroll
  for (int nt = 0; nt < 2; nt++) {
    int ng = n0 + nt * 16 + r;
    int bb = ng >> 12, nl = ng & 4095;
    #pragma unroll
    for (int mt = 0; mt < 2; mt++) {
      int ob = m0 + mt * 16 + half * 8;
      union { unsigned short s[8]; uint4 u; } pk;
      #pragma unroll
      for (int e = 0; e < 8; e++) {
        int o = ob + e;
        size_t xi = (size_t)bb * CCH * NN + (size_t)o * NN + nl;
        float y = acc[mt][nt][e] * s_p[o] + b_p[o] + x[xi];
        x1f[xi] = y;
        pk.s[e] = bfbits(y);
      }
      *(uint4*)(x1bf + (size_t)ng * CCH + ob) = pk.u;
    }
  }
}

// ---------------- MLP1: silu(W_m1 x1) -> bf16 [n][320] (rows >=307 are zero) ----------------
__global__ __launch_bounds__(256) void gemm_mlp1(const bf16* __restrict__ wm1,
                                                 const bf16* __restrict__ x1bf,
                                                 const float* __restrict__ s_m,
                                                 const float* __restrict__ b_m,
                                                 bf16* __restrict__ mbf) {
  GEMM_PROLOG
  gemm_core(wm1, x1bf, CCH, bm0, bn0, wave, lane, acc);
  int half = lane >> 4, r = lane & 15;
  #pragma unroll
  for (int nt = 0; nt < 2; nt++) {
    int ng = n0 + nt * 16 + r;
    #pragma unroll
    for (int mt = 0; mt < 2; mt++) {
      int ob = m0 + mt * 16 + half * 8;
      union { unsigned short s[8]; uint4 u; } pk;
      #pragma unroll
      for (int e = 0; e < 8; e++) {
        int o = ob + e;
        float y = 0.f;
        if (o < MD) {
          float v = acc[mt][nt][e] * s_m[o] + b_m[o];
          y = v / (1.f + __expf(-v));        // silu
        }
        pk.s[e] = bfbits(y);
      }
      *(uint4*)(mbf + (size_t)ng * MDP + ob) = pk.u;
    }
  }
}

// ---------------- MLP2: out = x1 + s*W_m2*m + b -> f32 [b][c][n] (d_out) ----------------
__global__ __launch_bounds__(256) void gemm_mlp2(const bf16* __restrict__ wm2,
                                                 const bf16* __restrict__ mbf,
                                                 const float* __restrict__ s_m,
                                                 const float* __restrict__ b_m,
                                                 const float* __restrict__ x1f,
                                                 float* __restrict__ outp) {
  GEMM_PROLOG
  gemm_core(wm2, mbf, MDP, bm0, bn0, wave, lane, acc);
  int half = lane >> 4, r = lane & 15;
  #pragma unroll
  for (int nt = 0; nt < 2; nt++) {
    int ng = n0 + nt * 16 + r;
    int bb = ng >> 12, nl = ng & 4095;
    #pragma unroll
    for (int mt = 0; mt < 2; mt++) {
      int ob = m0 + mt * 16 + half * 8;
      #pragma unroll
      for (int e = 0; e < 8; e++) {
        int o = ob + e;
        size_t xi = (size_t)bb * CCH * NN + (size_t)o * NN + nl;
        outp[xi] = acc[mt][nt][e] * s_m[o] + b_m[o] + x1f[xi];
      }
    }
  }
}

// ---------------- host launcher ----------------
extern "C" void kernel_launch(void* const* d_in, const int* in_sizes, int n_in,
                              void* d_out, int out_size, void* d_ws, size_t ws_size,
                              hipStream_t stream) {
  (void)in_sizes; (void)n_in; (void)out_size; (void)ws_size;
  const float* x      = (const float*)d_in[0];
  const float* w_qk   = (const float*)d_in[1];
  const float* s_qk   = (const float*)d_in[2];
  const float* b_qk   = (const float*)d_in[3];
  const float* w_v    = (const float*)d_in[4];
  const float* s_v    = (const float*)d_in[5];
  const float* b_v    = (const float*)d_in[6];
  const float* w_pe   = (const float*)d_in[7];
  const float* s_pe   = (const float*)d_in[8];
  const float* b_pe   = (const float*)d_in[9];
  const float* w_proj = (const float*)d_in[10];
  const float* s_proj = (const float*)d_in[11];
  const float* b_proj = (const float*)d_in[12];
  const float* w_m1   = (const float*)d_in[13];
  const float* s_m1   = (const float*)d_in[14];
  const float* b_m1   = (const float*)d_in[15];
  const float* w_m2   = (const float*)d_in[16];
  const float* s_m2   = (const float*)d_in[17];
  const float* b_m2   = (const float*)d_in[18];

  char* ws = (char*)d_ws;
  size_t off = 0;
  auto alloc = [&](size_t bytes) -> void* {
    void* p = ws + off; off += (bytes + 255) & ~(size_t)255; return p;
  };
  bf16* wqkv  = (bf16*)alloc((size_t)MQKV * CCH * 2);
  bf16* wproj = (bf16*)alloc((size_t)CCH * CCH * 2);
  bf16* wm1   = (bf16*)alloc((size_t)MDP * CCH * 2);
  bf16* wm2   = (bf16*)alloc((size_t)CCH * MDP * 2);
  bf16* xt    = (bf16*)alloc((size_t)NT * CCH * 2);
  bf16* qkt   = (bf16*)alloc((size_t)NT * QKC * 2);
  bf16* vbf   = (bf16*)alloc((size_t)BB * CCH * NN * 2);
  bf16* outbf = (bf16*)alloc((size_t)NT * CCH * 2);
  float* x1f  = (float*)alloc((size_t)NT * CCH * 4);
  bf16* x1bf  = (bf16*)alloc((size_t)NT * CCH * 2);
  bf16* mbf   = (bf16*)alloc((size_t)NT * MDP * 2);

  // weights -> bf16 (padded where needed)
  cvt_pad<<<(512 * 256 + 255) / 256, 256, 0, stream>>>(w_qk, wqkv, 512, 256, 512, 256);
  cvt_pad<<<(256 * 256 + 255) / 256, 256, 0, stream>>>(w_v, wqkv + 512 * 256, 256, 256, 256, 256);
  cvt_pad<<<(256 * 256 + 255) / 256, 256, 0, stream>>>(w_proj, wproj, 256, 256, 256, 256);
  cvt_pad<<<(MDP * 256 + 255) / 256, 256, 0, stream>>>(w_m1, wm1, MDP, 256, MD, 256);
  cvt_pad<<<(256 * MDP + 255) / 256, 256, 0, stream>>>(w_m2, wm2, 256, MDP, 256, MD);

  // x -> bf16 transposed [b][n][c]
  transpose_cvt<<<dim3(NN / 32, CCH / 32, BB), dim3(32, 8), 0, stream>>>(x, xt);

  // qkv fused GEMM (M=768, N=8192, K=256)
  gemm_qkv<<<dim3(NT / 128, MQKV / 64), 256, 0, stream>>>(wqkv, xt, s_qk, b_qk, s_v, b_v, qkt, vbf);

  // area attention (8 ba * 8 heads * 8 query-blocks of 128); 128KB dynamic LDS for K/V panels
  attn_kernel<<<512, 256, 131072, stream>>>(qkt, vbf, outbf);

  // depthwise 3x3 positional encoding, added in-place
  dwconv_pe<<<(BB * CCH * NN) / 256, 256, 0, stream>>>(vbf, w_pe, s_pe, b_pe, outbf);

  // projection + residual
  gemm_proj<<<dim3(NT / 128, CCH / 64), 256, 0, stream>>>(wproj, outbf, s_proj, b_proj, x, x1f, x1bf);

  // MLP
  gemm_mlp1<<<dim3(NT / 128, MDP / 64), 256, 0, stream>>>(wm1, x1bf, s_m1, b_m1, mbf);
  gemm_mlp2<<<dim3(NT / 128, CCH / 64), 256, 0, stream>>>(wm2, mbf, s_m2, b_m2, x1f, (float*)d_out);
}
